// MLA_44684839748001
// MI455X (gfx1250) — compile-verified, matrix operand order checked by reading, not run on hardware
//
#include <hip/hip_runtime.h>

// ---------------------------------------------------------------------------
// MLA forward for gfx1250 (MI455X): bf16 WMMA everywhere, fp32 accumulation.
// GEMMs + attention: LDS-staged, double-buffered, async-to-LDS (ASYNCcnt).
// ---------------------------------------------------------------------------

typedef __bf16 bf16;
typedef __attribute__((ext_vector_type(8)))  __bf16 v8bf;
typedef __attribute__((ext_vector_type(16))) __bf16 v16bf;
typedef __attribute__((ext_vector_type(8)))  float  v8f;
typedef __attribute__((ext_vector_type(4)))  int    v4i;

#define GLOBAL_AS __attribute__((address_space(1)))
#define LDS_AS    __attribute__((address_space(3)))

#define SEQ   2048
#define HID   2048
#define NHEAD 16
#define QLR   1536
#define KVLR  512
#define ROPED 64
#define NOPE  128
#define VDIM  128
#define QHD   192            // NOPE + ROPED
#define KVUPN 4096           // NHEAD*(NOPE+VDIM)
#define QUPN  3072           // NHEAD*QHD
#define KVDN  576            // KVLR + ROPED

#if defined(__has_builtin)
#  if __has_builtin(__builtin_amdgcn_global_load_async_to_lds_b128)
#    define HAVE_ASYNC_LDS 1
#  endif
#endif
#ifndef HAVE_ASYNC_LDS
#  define HAVE_ASYNC_LDS 0
#endif

// ---------------------------------------------------------------------------
// Fragment helpers.
// A-fragment (16x32, bf16): lane L holds row m = L%16; K elements
//   k = (L/16)*8 + {0..7} and (L/16)*8 + 16 + {0..7}  (two 16B contiguous runs)
// B-fragment (32x16): lane L holds column n = L%16 with the same K pattern,
//   i.e. contiguous reads from W stored (N,K) row-major.
// C/D-fragment (16x16 f32, 8 VGPRs): elem i at lane L -> (M = i + 8*(L/16),
//   N = L%16).
// ---------------------------------------------------------------------------
__device__ __forceinline__ v16bf mla_load_frag(const bf16* rowptr, int k0,
                                               int half) {
  const v8bf a = *(const v8bf*)(rowptr + k0 + half * 8);
  const v8bf b = *(const v8bf*)(rowptr + k0 + half * 8 + 16);
  v16bf f;
#pragma unroll
  for (int i = 0; i < 8; ++i) { f[i] = a[i]; f[i + 8] = b[i]; }
  return f;
}

__device__ __forceinline__ v8f mla_wmma(v16bf a, v16bf b, v8f c) {
  return __builtin_amdgcn_wmma_f32_16x16x32_bf16(
      /*neg_a=*/false, a, /*neg_b=*/false, b,
      /*c_mod=*/(short)0, c, /*reuse_a=*/false, /*reuse_b=*/false);
}

// 16-byte global -> LDS copy: async (ASYNCcnt-tracked) when available.
__device__ __forceinline__ void mla_cp16(const bf16* gsrc, bf16* ldst) {
#if HAVE_ASYNC_LDS
  __builtin_amdgcn_global_load_async_to_lds_b128(
      (GLOBAL_AS v4i*)(void*)gsrc, (LDS_AS v4i*)(void*)ldst,
      /*offset=*/0, /*cpol=*/0);
#else
  *(v8bf*)ldst = *(const v8bf*)gsrc;
#endif
}

template <int N>
__device__ __forceinline__ void mla_wait_async() {
#if HAVE_ASYNC_LDS
  asm volatile("s_wait_asynccnt %0" ::"i"(N) : "memory");
#endif
}

// ---------------------------------------------------------------------------
// fp32 -> bf16 conversion (grid-stride)
// ---------------------------------------------------------------------------
__global__ void mla_cvt_f32_bf16(const float* __restrict__ in,
                                 bf16* __restrict__ out, long n) {
  long i = (long)blockIdx.x * blockDim.x + threadIdx.x;
  long stride = (long)gridDim.x * blockDim.x;
  for (; i < n; i += stride) out[i] = (bf16)in[i];
}

// ---------------------------------------------------------------------------
// WMMA GEMM:  C[M,N] = A[M,K](bf16) * W[N,K]^T(bf16) + bias (+ resid)
// Block: 128 threads = 4 waves. Block tile 128x64; wave tile 32x64.
// K processed in 64-wide stages, double-buffered through LDS via async copy.
// Requires: M % 128 == 0, N % 64 == 0, K % 64 == 0.
// ---------------------------------------------------------------------------
__global__ __launch_bounds__(128)
void mla_gemm_bf16(const bf16* __restrict__ A, const bf16* __restrict__ W,
                   const float* __restrict__ bias,
                   const float* __restrict__ resid,
                   float* __restrict__ C, int M, int N, int K) {
  __shared__ __align__(16) bf16 Asm[2][128 * 64];
  __shared__ __align__(16) bf16 Bsm[2][64 * 64];

  const int tid  = threadIdx.x;
  const int lane = tid & 31;
  const int wave = tid >> 5;
  const int half = lane >> 4;
  const int l16  = lane & 15;
  const int nbase = blockIdx.x * 64;
  const int mtile = blockIdx.y * 128;
  const int nk = K >> 6;

  // Cooperative copy of one K-stage (A: 128x64, B: 64x64) into buffer `buf`.
  auto stage = [&](int kt, int buf) {
    const bf16* ga = A + (size_t)mtile * K + kt * 64;
#pragma unroll
    for (int i = 0; i < 8; ++i) {
      const int chunk = i * 128 + tid;    // 0..1023 chunks of 16B
      const int row = chunk >> 3, c = chunk & 7;
      mla_cp16(ga + (size_t)row * K + c * 8, &Asm[buf][row * 64 + c * 8]);
    }
    const bf16* gb = W + (size_t)nbase * K + kt * 64;
#pragma unroll
    for (int i = 0; i < 4; ++i) {
      const int chunk = i * 128 + tid;    // 0..511
      const int row = chunk >> 3, c = chunk & 7;
      mla_cp16(gb + (size_t)row * K + c * 8, &Bsm[buf][row * 64 + c * 8]);
    }
  };

  v8f acc[2][4] = {};
  stage(0, 0);
  if (nk > 1) stage(1, 1);

  for (int kt = 0; kt < nk; ++kt) {
    const int buf = kt & 1;
    // Leave the next stage (12 async ops/thread) in flight; drain this one.
    if (kt + 1 < nk) mla_wait_async<12>();
    else             mla_wait_async<0>();
    __syncthreads();

    const bf16* arow0 = &Asm[buf][(wave * 32 + l16) * 64];
    const bf16* arow1 = arow0 + 16 * 64;
    const bf16* brow  = &Bsm[buf][l16 * 64];
#pragma unroll
    for (int ks = 0; ks < 64; ks += 32) {
      v16bf a0 = mla_load_frag(arow0, ks, half);
      v16bf a1 = mla_load_frag(arow1, ks, half);
#pragma unroll
      for (int j = 0; j < 4; ++j) {
        v16bf bfrag = mla_load_frag(brow + j * 16 * 64, ks, half);
        acc[0][j] = mla_wmma(a0, bfrag, acc[0][j]);
        acc[1][j] = mla_wmma(a1, bfrag, acc[1][j]);
      }
    }
    __syncthreads();
    if (kt + 2 < nk) stage(kt + 2, buf);
  }

#pragma unroll
  for (int r = 0; r < 2; ++r) {
#pragma unroll
    for (int j = 0; j < 4; ++j) {
      const int col = nbase + j * 16 + l16;
      const float b = bias ? bias[col] : 0.0f;
#pragma unroll
      for (int i = 0; i < 8; ++i) {
        const int row = mtile + wave * 32 + r * 16 + i + 8 * half;
        float v = acc[r][j][i] + b;
        if (resid) v += resid[(size_t)row * N + col];
        C[(size_t)row * N + col] = v;
      }
    }
  }
}

// ---------------------------------------------------------------------------
// RMS norm over n elements of each row (row stride in_stride), bf16 output.
// ---------------------------------------------------------------------------
__global__ __launch_bounds__(256)
void mla_rmsnorm_bf16(const float* __restrict__ X, const float* __restrict__ w,
                      bf16* __restrict__ Y, int n, int in_stride) {
  __shared__ float red[8];
  const int row = blockIdx.x;
  const float* x = X + (size_t)row * in_stride;
  float ss = 0.0f;
  for (int i = threadIdx.x; i < n; i += blockDim.x) {
    float v = x[i];
    ss += v * v;
  }
#pragma unroll
  for (int off = 16; off > 0; off >>= 1) ss += __shfl_down(ss, off);
  if ((threadIdx.x & 31) == 0) red[threadIdx.x >> 5] = ss;
  __syncthreads();
  if (threadIdx.x < 8) {
    float v = red[threadIdx.x];
#pragma unroll
    for (int off = 4; off > 0; off >>= 1) v += __shfl_down(v, off);
    if (threadIdx.x == 0) red[0] = v;
  }
  __syncthreads();
  const float inv = rsqrtf(red[0] / (float)n + 1.1920929e-07f);
  bf16* y = Y + (size_t)row * n;
  for (int i = threadIdx.x; i < n; i += blockDim.x)
    y[i] = (bf16)(x[i] * inv * w[i]);
}

// ---------------------------------------------------------------------------
// RoPE on Q + repack to [NHEAD][SEQ][QHD] bf16, pre-scaled by 1/sqrt(QHD).
// ---------------------------------------------------------------------------
__global__ __launch_bounds__(192)
void mla_rope_pack_q(const float* __restrict__ Qf,
                     const int* __restrict__ pos_ids,
                     bf16* __restrict__ Qb) {
  const int h = blockIdx.x, s = blockIdx.y, j = threadIdx.x;
  const float* q = Qf + (size_t)s * QUPN + (size_t)h * QHD;
  const float scale = 0.07216878364870323f;  // 1/sqrt(192)
  float v;
  if (j < NOPE) {
    v = q[j];
  } else {
    const int jr = j - NOPE;            // 0..63
    const int fi = jr & 31;
    const float pos = (float)pos_ids[s];
    const float invf = __powf(10000.0f, -(float)fi / 32.0f);
    const float ang = pos * invf;
    const float x = q[NOPE + jr];
    const float rot = (jr < 32) ? -q[NOPE + jr + 32] : q[NOPE + jr - 32];
    v = x * __cosf(ang) + rot * __sinf(ang);
  }
  Qb[((size_t)h * SEQ + s) * QHD + j] = (bf16)(v * scale);
}

// ---------------------------------------------------------------------------
// RoPE on shared k_rope (cols 512..575 of c_kv), broadcast to all heads into
// Kb[h][s][128..191].
// ---------------------------------------------------------------------------
__global__ __launch_bounds__(64)
void mla_rope_pack_k(const float* __restrict__ CKVf,
                     const int* __restrict__ pos_ids,
                     bf16* __restrict__ Kb) {
  const int s = blockIdx.x, jr = threadIdx.x;  // 0..63
  const float* kr = CKVf + (size_t)s * KVDN + KVLR;
  const int fi = jr & 31;
  const float pos = (float)pos_ids[s];
  const float invf = __powf(10000.0f, -(float)fi / 32.0f);
  const float ang = pos * invf;
  const float x = kr[jr];
  const float rot = (jr < 32) ? -kr[jr + 32] : kr[jr - 32];
  const bf16 v = (bf16)(x * __cosf(ang) + rot * __sinf(ang));
#pragma unroll
  for (int h = 0; h < NHEAD; ++h)
    Kb[((size_t)h * SEQ + s) * QHD + NOPE + jr] = v;
}

// ---------------------------------------------------------------------------
// Split kv_up output: k_nope -> Kb[h][s][0..127]; V -> Vt[h][d][s] (transposed
// so P*V B-fragments become contiguous loads).
// ---------------------------------------------------------------------------
__global__ __launch_bounds__(256)
void mla_pack_kv(const float* __restrict__ KVf, bf16* __restrict__ Kb,
                 bf16* __restrict__ Vt) {
  const int h = blockIdx.x, s = blockIdx.y, j = threadIdx.x;
  const float* kv = KVf + (size_t)s * KVUPN + (size_t)h * (NOPE + VDIM);
  if (j < NOPE) {
    Kb[((size_t)h * SEQ + s) * QHD + j] = (bf16)kv[j];
  } else {
    const int d = j - NOPE;
    Vt[((size_t)h * VDIM + d) * SEQ + s] = (bf16)kv[NOPE + d];
  }
}

// ---------------------------------------------------------------------------
// Flash attention, block = 4 waves sharing one head; each wave owns a 16-row
// Q tile (q0 = blockIdx.y*64 + ty*16). K (32x192) and V (128x32) tiles are
// staged once per block into LDS via double-buffered async copies and shared
// by all 4 waves. All waves iterate the block-uniform key range
// [0, q0_block+64) (causal) so block barriers are safe; fully-masked tiles
// contribute 0 through the online softmax.
// ---------------------------------------------------------------------------
__global__ __launch_bounds__(128)
void mla_flash_attn(const bf16* __restrict__ Qb, const bf16* __restrict__ Kb,
                    const bf16* __restrict__ Vt, bf16* __restrict__ Obf,
                    const int* __restrict__ maskedp) {
  __shared__ __align__(16) bf16  Ksm[2][32 * QHD];   // 2 x 12KB
  __shared__ __align__(16) bf16  Vsm[2][VDIM * 32];  // 2 x 8KB, [d][k]
  __shared__ __align__(16) float Sld[4][16 * 32];    // 8KB
  __shared__ __align__(16) bf16  Pld[4][16 * 32];    // 4KB

  const int tid  = threadIdx.x;
  const int lane = tid & 31;
  const int ty   = tid >> 5;
  const int half = lane >> 4;
  const int l16  = lane & 15;
  const int h    = blockIdx.x;
  const int q0b  = blockIdx.y * 64;
  const int q0   = q0b + ty * 16;
  const bool causal = (*maskedp) != 0;

  // Q fragments resident in registers (16 rows x 192).
  const bf16* qrow = Qb + ((size_t)h * SEQ + q0 + l16) * QHD;
  v16bf qf[6];
#pragma unroll
  for (int c = 0; c < 6; ++c) qf[c] = mla_load_frag(qrow, c * 32, half);

  // Stage one 32-key tile: K rows kb..kb+31 (32x192) + V cols kb..kb+31
  // (128x32, from transposed Vt). 10 async 16B ops per thread.
  auto stage = [&](int kb, int buf) {
    const bf16* gk = Kb + ((size_t)h * SEQ + kb) * QHD;
#pragma unroll
    for (int i = 0; i < 6; ++i) {
      const int chunk = i * 128 + tid;    // 0..767 (32 rows x 24 chunks)
      const int row = chunk / 24, c = chunk % 24;
      mla_cp16(gk + (size_t)row * QHD + c * 8, &Ksm[buf][row * QHD + c * 8]);
    }
    const bf16* gv = Vt + (size_t)h * VDIM * SEQ + kb;
#pragma unroll
    for (int i = 0; i < 4; ++i) {
      const int chunk = i * 128 + tid;    // 0..511 (128 rows x 4 chunks)
      const int row = chunk >> 2, c = chunk & 3;
      mla_cp16(gv + (size_t)row * SEQ + c * 8, &Vsm[buf][row * 32 + c * 8]);
    }
  };

  v8f o[8] = {};
  float m_i = -3.0e38f, l_i = 0.0f;

  const int kend = causal ? (q0b + 64) : SEQ;  // block-uniform
  const int nkt = kend >> 5;                   // 32-key tiles

  stage(0, 0);
  if (nkt > 1) stage(32, 1);

  for (int kt = 0; kt < nkt; ++kt) {
    const int kb = kt * 32;
    const int buf = kt & 1;
    if (kt + 1 < nkt) mla_wait_async<10>();
    else              mla_wait_async<0>();
    __syncthreads();

    // ---- S = Q * K^T (two 16x16 tiles covering 32 key columns) ----
    v8f s[2] = {};
#pragma unroll
    for (int j = 0; j < 2; ++j) {
      const bf16* krow = &Ksm[buf][(j * 16 + l16) * QHD];
#pragma unroll
      for (int c = 0; c < 6; ++c) {
        v16bf kf = mla_load_frag(krow, c * 32, half);
        s[j] = mla_wmma(qf[c], kf, s[j]);
      }
    }

    // ---- causal mask + stage S into per-wave LDS (row-major 16x32) ----
#pragma unroll
    for (int j = 0; j < 2; ++j) {
      const int col = kb + j * 16 + l16;
#pragma unroll
      for (int i = 0; i < 8; ++i) {
        const int row = q0 + i + 8 * half;
        float v = s[j][i];
        if (causal && col > row) v = -3.0e38f;
        Sld[ty][(i + 8 * half) * 32 + j * 16 + l16] = v;
      }
    }
    asm volatile("s_wait_dscnt 0x0" ::: "memory");

    // ---- online softmax: lanes 0..15 each own one row ----
    float alpha = 1.0f;
    if (lane < 16) {
      const float* srow = &Sld[ty][lane * 32];
      float tmax = -3.0e38f;
#pragma unroll
      for (int j = 0; j < 32; ++j) tmax = fmaxf(tmax, srow[j]);
      const float mnew = fmaxf(m_i, tmax);
      alpha = __expf(m_i - mnew);
      float lsum = 0.0f;
      bf16* prow = &Pld[ty][lane * 32];
#pragma unroll
      for (int j = 0; j < 32; ++j) {
        const float p = __expf(srow[j] - mnew);
        lsum += p;
        prow[j] = (bf16)p;
      }
      l_i = l_i * alpha + lsum;
      m_i = mnew;
    }
    asm volatile("s_wait_dscnt 0x0" ::: "memory");

    // ---- rescale O accumulators by alpha (broadcast per row) ----
#pragma unroll
    for (int i = 0; i < 8; ++i) {
      const float a = __shfl(alpha, i + 8 * half);
#pragma unroll
      for (int j = 0; j < 8; ++j) o[j][i] *= a;
    }

    // ---- load P as A-fragment from per-wave LDS ----
    v16bf pf;
    {
      const bf16* pp = &Pld[ty][l16 * 32 + half * 8];
      const v8bf p0 = *(const v8bf*)(pp);
      const v8bf p1 = *(const v8bf*)(pp + 16);
#pragma unroll
      for (int i = 0; i < 8; ++i) { pf[i] = p0[i]; pf[i + 8] = p1[i]; }
    }

    // ---- O += P * V from the shared V tile ----
#pragma unroll
    for (int j = 0; j < 8; ++j) {
      const bf16* vrow = &Vsm[buf][(j * 16 + l16) * 32];
      v16bf vf = mla_load_frag(vrow, 0, half);
      o[j] = mla_wmma(pf, vf, o[j]);
    }

    __syncthreads();
    if (kt + 2 < nkt) stage(kb + 64, buf);
  }

  // ---- finalize: divide by row sums, store bf16 ----
#pragma unroll
  for (int i = 0; i < 8; ++i) {
    const float linv = 1.0f / __shfl(l_i, i + 8 * half);
    const int row = q0 + i + 8 * half;
#pragma unroll
    for (int j = 0; j < 8; ++j) {
      const int col = h * VDIM + j * 16 + l16;
      Obf[(size_t)row * (NHEAD * VDIM) + col] = (bf16)(o[j][i] * linv);
    }
  }
}

// ---------------------------------------------------------------------------
// Host-side orchestration.
// ---------------------------------------------------------------------------
static inline char* ws_take(char*& p, size_t bytes) {
  char* r = p;
  p += (bytes + 255) & ~(size_t)255;
  return r;
}

extern "C" void kernel_launch(void* const* d_in, const int* in_sizes, int n_in,
                              void* d_out, int out_size, void* d_ws,
                              size_t ws_size, hipStream_t stream) {
  (void)in_sizes; (void)n_in; (void)out_size; (void)ws_size;

  const float* hidden   = (const float*)d_in[0];
  const int*   pos_ids  = (const int*)d_in[1];
  const int*   maskedp  = (const int*)d_in[2];
  const float* Wq_down  = (const float*)d_in[3];
  const float* bq_down  = (const float*)d_in[4];
  const float* q_norm_w = (const float*)d_in[5];
  const float* Wq_up    = (const float*)d_in[6];
  const float* bq_up    = (const float*)d_in[7];
  const float* Wkv_down = (const float*)d_in[8];
  const float* bkv_down = (const float*)d_in[9];
  const float* kv_norm_w= (const float*)d_in[10];
  const float* Wkv_up   = (const float*)d_in[11];
  const float* bkv_up   = (const float*)d_in[12];
  const float* Wo       = (const float*)d_in[13];
  const float* bo       = (const float*)d_in[14];
  float* out = (float*)d_out;

  char* wp = (char*)d_ws;
  bf16* hbf     = (bf16*)ws_take(wp, (size_t)SEQ * HID * 2);
  bf16* wqd_bf  = (bf16*)ws_take(wp, (size_t)QLR * HID * 2);
  bf16* wqu_bf  = (bf16*)ws_take(wp, (size_t)QUPN * QLR * 2);
  bf16* wkvd_bf = (bf16*)ws_take(wp, (size_t)KVDN * HID * 2);
  bf16* wkvu_bf = (bf16*)ws_take(wp, (size_t)KVUPN * KVLR * 2);
  bf16* wo_bf   = (bf16*)ws_take(wp, (size_t)HID * (NHEAD * VDIM) * 2);
  float* cq_f   = (float*)ws_take(wp, (size_t)SEQ * QLR * 4);
  bf16* qdn_bf  = (bf16*)ws_take(wp, (size_t)SEQ * QLR * 2);
  float* q_f    = (float*)ws_take(wp, (size_t)SEQ * QUPN * 4);
  float* ckv_f  = (float*)ws_take(wp, (size_t)SEQ * KVDN * 4);
  bf16* ckv_bf  = (bf16*)ws_take(wp, (size_t)SEQ * KVLR * 2);
  float* kv_f   = (float*)ws_take(wp, (size_t)SEQ * KVUPN * 4);
  bf16* Qb      = (bf16*)ws_take(wp, (size_t)NHEAD * SEQ * QHD * 2);
  bf16* Kb      = (bf16*)ws_take(wp, (size_t)NHEAD * SEQ * QHD * 2);
  bf16* Vt      = (bf16*)ws_take(wp, (size_t)NHEAD * VDIM * SEQ * 2);
  bf16* attn_bf = (bf16*)ws_take(wp, (size_t)SEQ * (NHEAD * VDIM) * 2);

  // 1. fp32 -> bf16 conversions
  auto cvt = [&](const float* src, bf16* dst, long n) {
    int grid = (int)((n + 255) / 256);
    if (grid > 2048) grid = 2048;
    mla_cvt_f32_bf16<<<grid, 256, 0, stream>>>(src, dst, n);
  };
  cvt(hidden,   hbf,     (long)SEQ * HID);
  cvt(Wq_down,  wqd_bf,  (long)QLR * HID);
  cvt(Wq_up,    wqu_bf,  (long)QUPN * QLR);
  cvt(Wkv_down, wkvd_bf, (long)KVDN * HID);
  cvt(Wkv_up,   wkvu_bf, (long)KVUPN * KVLR);
  cvt(Wo,       wo_bf,   (long)HID * (NHEAD * VDIM));

  // 2. down projections
  mla_gemm_bf16<<<dim3(QLR / 64, SEQ / 128), 128, 0, stream>>>(
      hbf, wqd_bf, bq_down, nullptr, cq_f, SEQ, QLR, HID);
  mla_gemm_bf16<<<dim3(KVDN / 64, SEQ / 128), 128, 0, stream>>>(
      hbf, wkvd_bf, bkv_down, nullptr, ckv_f, SEQ, KVDN, HID);

  // 3. RMS norms -> bf16
  mla_rmsnorm_bf16<<<SEQ, 256, 0, stream>>>(cq_f, q_norm_w, qdn_bf, QLR, QLR);
  mla_rmsnorm_bf16<<<SEQ, 256, 0, stream>>>(ckv_f, kv_norm_w, ckv_bf, KVLR, KVDN);

  // 4. up projections
  mla_gemm_bf16<<<dim3(QUPN / 64, SEQ / 128), 128, 0, stream>>>(
      qdn_bf, wqu_bf, bq_up, nullptr, q_f, SEQ, QUPN, QLR);
  mla_gemm_bf16<<<dim3(KVUPN / 64, SEQ / 128), 128, 0, stream>>>(
      ckv_bf, wkvu_bf, bkv_up, nullptr, kv_f, SEQ, KVUPN, KVLR);

  // 5. RoPE + repack
  mla_rope_pack_q<<<dim3(NHEAD, SEQ), 192, 0, stream>>>(q_f, pos_ids, Qb);
  mla_rope_pack_k<<<SEQ, 64, 0, stream>>>(ckv_f, pos_ids, Kb);
  mla_pack_kv<<<dim3(NHEAD, SEQ), 256, 0, stream>>>(kv_f, Kb, Vt);

  // 6. flash attention
  mla_flash_attn<<<dim3(NHEAD, SEQ / 64), 128, 0, stream>>>(
      Qb, Kb, Vt, attn_bf, maskedp);

  // 7. output projection + bias + residual
  mla_gemm_bf16<<<dim3(HID / 64, SEQ / 128), 128, 0, stream>>>(
      attn_bf, wo_bf, bo, hidden, out, SEQ, HID, NHEAD * VDIM);
}
